// ModalityGraphAttention_51264729645567
// MI455X (gfx1250) — matrix operand order repaired and checked by reading, not compiled, not run on hardware
//
#include <hip/hip_runtime.h>
#include <stdint.h>

#define B_  8
#define N_  1024
#define D_  512
#define H_  8
#define HD_ 64

typedef __bf16 bf16_t;
typedef __attribute__((ext_vector_type(16))) __bf16 v16bf;
typedef __attribute__((ext_vector_type(8)))  float  v8f;

struct Bytes32 { uint4 lo, hi; };

static __device__ __forceinline__ v16bf frag_from(uint4 lo, uint4 hi) {
  Bytes32 u; u.lo = lo; u.hi = hi;
  return __builtin_bit_cast(v16bf, u);
}

// A-matrix fragment (16x32 bf16), A row-major, leading dim lda (elements).
// ISA layout: lanes 0-15 row=lane, K 0-7 (v0-3) & 16-23 (v4-7);
//             lanes 16-31 row=lane-16, K 8-15 & 24-31.
static __device__ __forceinline__ v16bf load_a_frag(const bf16_t* A, int lda) {
  const int l   = threadIdx.x & 31;
  const int row = l & 15;
  const int kb  = (l >> 4) << 3;          // 0 or 8
  const bf16_t* p = A + row * lda + kb;
  uint4 lo = *(const uint4*)(p);          // K kb..kb+7
  uint4 hi = *(const uint4*)(p + 16);     // K kb+16..kb+23
  return frag_from(lo, hi);
}

// B-matrix fragment (32x16 bf16) from Bt = B transposed (N x K row-major).
// ISA layout: lanes 0-15 col=lane, K 0-7 & 16-23; lanes 16-31 col=lane-16, K 8-15 & 24-31.
static __device__ __forceinline__ v16bf load_b_frag(const bf16_t* Bt, int ldb) {
  const int l   = threadIdx.x & 31;
  const int col = l & 15;
  const int kg  = (l >> 4) << 3;          // 0 or 8
  const bf16_t* p = Bt + col * ldb + kg;
  uint4 lo = *(const uint4*)(p);
  uint4 hi = *(const uint4*)(p + 16);
  return frag_from(lo, hi);
}

static __device__ __forceinline__ bf16_t f2bf(float f) {
  uint32_t u = __builtin_bit_cast(uint32_t, f);
  u += 0x7FFFu + ((u >> 16) & 1u);        // round-to-nearest-even
  uint16_t h = (uint16_t)(u >> 16);
  return __builtin_bit_cast(bf16_t, h);
}

static __device__ __forceinline__ v8f v8zero() {
  v8f z;
#pragma unroll
  for (int i = 0; i < 8; ++i) z[i] = 0.0f;
  return z;
}

static __device__ __forceinline__ float redmax16(float v) {
  v = fmaxf(v, __shfl_xor(v, 1, 32));
  v = fmaxf(v, __shfl_xor(v, 2, 32));
  v = fmaxf(v, __shfl_xor(v, 4, 32));
  v = fmaxf(v, __shfl_xor(v, 8, 32));
  return v;
}
static __device__ __forceinline__ float redsum16(float v) {
  v += __shfl_xor(v, 1, 32);
  v += __shfl_xor(v, 2, 32);
  v += __shfl_xor(v, 4, 32);
  v += __shfl_xor(v, 8, 32);
  return v;
}

#define WMMA_BF16(a, b, c) \
  __builtin_amdgcn_wmma_f32_16x16x32_bf16(false, (a), false, (b), (short)0, (c), false, false)

// Async copy of one 16B chunk global -> LDS (gfx1250 GLOBAL_LOAD_ASYNC_TO_LDS_B128).
static __device__ __forceinline__ void async_g2l_b128(uint32_t lds_off, uint64_t gaddr) {
  asm volatile("global_load_async_to_lds_b128 %0, %1, off"
               :: "v"(lds_off), "v"(gaddr) : "memory");
}
static __device__ __forceinline__ void wait_async0() {
  asm volatile("s_wait_asynccnt 0x0" ::: "memory");
}

// ---------------- kernel 1: f32 -> bf16 conversion ----------------
__global__ void k_cvt_bf16(const float* __restrict__ src, bf16_t* __restrict__ dst, int n) {
  int i = (blockIdx.x * blockDim.x + threadIdx.x) * 4;
  if (i + 3 < n) {
    float4 v = *(const float4*)(src + i);
    dst[i + 0] = f2bf(v.x);
    dst[i + 1] = f2bf(v.y);
    dst[i + 2] = f2bf(v.z);
    dst[i + 3] = f2bf(v.w);
  }
}

// ---------------- kernel 2: QKV projection (32x64 tile per wave) ----------------
// Q,K: [B,H,N,HD] bf16 row-major; V stored transposed Vt: [B,H,HD,N] bf16.
__global__ void k_qkv(const bf16_t* __restrict__ xb,
                      const bf16_t* __restrict__ wq,
                      const bf16_t* __restrict__ wk,
                      const bf16_t* __restrict__ wv,
                      const float* __restrict__ bq,
                      const float* __restrict__ bk,
                      const float* __restrict__ bv,
                      bf16_t* __restrict__ Q,
                      bf16_t* __restrict__ K,
                      bf16_t* __restrict__ Vt) {
  const int w = blockIdx.x * 8 + (threadIdx.x >> 5);
  const int mat   = w >> 11;                // 0=Q 1=K 2=V (2048 waves per matrix)
  const int rem   = w & 2047;
  const int mbase = (rem >> 3) << 5;        // 32-row tile over B*N
  const int nbase = (rem & 7) << 6;         // 64-wide col tile over D

  const bf16_t* W    = (mat == 0) ? wq : (mat == 1) ? wk : wv;
  const float*  bias = (mat == 0) ? bq : (mat == 1) ? bk : bv;

  v8f acc0[4], acc1[4];
#pragma unroll
  for (int s = 0; s < 4; ++s) { acc0[s] = v8zero(); acc1[s] = v8zero(); }

  for (int k = 0; k < D_; k += 32) {
    __builtin_prefetch(xb + (size_t)mbase * D_ + k + 32, 0, 0);
    v16bf a0 = load_a_frag(xb + (size_t)mbase * D_ + k, D_);
    v16bf a1 = load_a_frag(xb + (size_t)(mbase + 16) * D_ + k, D_);
#pragma unroll
    for (int s = 0; s < 4; ++s) {
      v16bf b = load_b_frag(W + (size_t)(nbase + s * 16) * D_ + k, D_);
      acc0[s] = WMMA_BF16(a0, b, acc0[s]);
      acc1[s] = WMMA_BF16(a1, b, acc1[s]);
    }
  }

  const int l    = threadIdx.x & 31;
  const int colL = l & 15;
  const int rb   = (l >> 4) << 3;
#pragma unroll
  for (int half = 0; half < 2; ++half) {
#pragma unroll
    for (int s = 0; s < 4; ++s) {
      const int n = nbase + s * 16 + colL;
      const int h = n >> 6, d = n & 63;
      const float bn = bias[n];
#pragma unroll
      for (int r = 0; r < 8; ++r) {
        const int m  = mbase + half * 16 + rb + r;
        const int bb = m >> 10, nn = m & 1023;
        const float val = (half ? acc1[s][r] : acc0[s][r]) + bn;
        if (mat == 0)
          Q[(((size_t)bb * H_ + h) * N_ + nn) * HD_ + d] = f2bf(val);
        else if (mat == 1)
          K[(((size_t)bb * H_ + h) * N_ + nn) * HD_ + d] = f2bf(val);
        else
          Vt[(((size_t)bb * H_ + h) * HD_ + d) * N_ + nn] = f2bf(val);
      }
    }
  }
}

// ---------------- kernel 3: flash attention with multiplicative mask ----------------
// One wave per (b, h, 16-row tile); all 8 waves of a block share (b,h) and stage
// the K/V j-tiles into LDS once per block via async global->LDS copies.
__global__ void k_attn(const bf16_t* __restrict__ Q,
                       const bf16_t* __restrict__ K,
                       const bf16_t* __restrict__ Vt,
                       const float* __restrict__ adj,
                       bf16_t* __restrict__ Out) {
  __shared__ bf16_t Ktile[32][64];      // keys j..j+31, 4KB
  __shared__ bf16_t Vtile[64][32];      // Vt[0:64][j..j+31], 4KB
  __shared__ bf16_t lds_p[8][16][32];   // per-wave P scratch (C-layout -> A-layout), 8KB
  const int t  = threadIdx.x;           // 0..255
  const int wl = t >> 5;
  const int w  = blockIdx.x * 8 + wl;
  const int bh    = w >> 6;             // same for all waves in a block
  const int mtile = w & 63;
  const int b = bh >> 3, h = bh & 7;

  const bf16_t* Qp   = Q  + (size_t)bh * N_ * HD_;
  const bf16_t* Kp   = K  + (size_t)bh * N_ * HD_;
  const bf16_t* Vtp  = Vt + (size_t)bh * HD_ * N_;
  const float*  adjp = adj + (size_t)b * N_ * N_ + (size_t)(mtile * 16) * N_;

  const int l    = t & 31;
  const int colL = l & 15;
  const int rb   = (l >> 4) << 3;

  v16bf qa0 = load_a_frag(Qp + (size_t)(mtile * 16) * HD_ + 0,  HD_);
  v16bf qa1 = load_a_frag(Qp + (size_t)(mtile * 16) * HD_ + 32, HD_);

  float mrow[8], lrow[8];
  v8f acc[4];
#pragma unroll
  for (int r = 0; r < 8; ++r) { mrow[r] = -3.0e38f; lrow[r] = 0.0f; }
#pragma unroll
  for (int dt = 0; dt < 4; ++dt) acc[dt] = v8zero();

  bf16_t* pl = &lds_p[wl][0][0];
  const float scale = 0.125f;   // 1/sqrt(64)

  // Per-thread async-copy assignments (16B each, 256 threads x 16B = 4KB per tile).
  const uint32_t ldsK = (uint32_t)(size_t)(&Ktile[0][0]) + (uint32_t)t * 16u;
  const uint32_t ldsV = (uint32_t)(size_t)(&Vtile[0][0]) + (uint32_t)t * 16u;
  const bf16_t* gVrow = Vtp + (size_t)(t >> 2) * N_ + (size_t)(t & 3) * 8;

#pragma unroll 1
  for (int j = 0; j < N_; j += 32) {
    __syncthreads();   // previous iteration's LDS reads complete
    // K tile rows j..j+31 are one contiguous 4KB block (row-major [N,HD]).
    async_g2l_b128(ldsK, (uint64_t)(size_t)(Kp + (size_t)j * HD_) + (uint64_t)t * 16u);
    // V tile: 64 rows x 32 cols; 4 threads per row, 16B segments.
    async_g2l_b128(ldsV, (uint64_t)(size_t)(gVrow + j));
    wait_async0();
    __syncthreads();   // all waves' tiles visible

    v8f s0 = v8zero(), s1 = v8zero();
    {
      v16bf kb;
      kb = load_b_frag(&Ktile[0][0], HD_);
      s0 = WMMA_BF16(qa0, kb, s0);
      kb = load_b_frag(&Ktile[0][0] + 32, HD_);
      s0 = WMMA_BF16(qa1, kb, s0);
      kb = load_b_frag(&Ktile[16][0], HD_);
      s1 = WMMA_BF16(qa0, kb, s1);
      kb = load_b_frag(&Ktile[16][0] + 32, HD_);
      s1 = WMMA_BF16(qa1, kb, s1);
    }
    float p0[8], p1[8], alpha[8];
#pragma unroll
    for (int r = 0; r < 8; ++r) {
      const float a0 = adjp[(size_t)(rb + r) * N_ + j + colL];
      const float a1 = adjp[(size_t)(rb + r) * N_ + j + 16 + colL];
      float v0 = s0[r] * scale * a0;     // masked entries become exactly 0 (per reference)
      float v1 = s1[r] * scale * a1;
      float tm = redmax16(fmaxf(v0, v1));
      float mn = fmaxf(mrow[r], tm);
      alpha[r] = __expf(mrow[r] - mn);
      v0 = __expf(v0 - mn);
      v1 = __expf(v1 - mn);
      float rs = redsum16(v0 + v1);
      lrow[r] = lrow[r] * alpha[r] + rs;
      mrow[r] = mn;
      p0[r] = v0; p1[r] = v1;
    }
#pragma unroll
    for (int dt = 0; dt < 4; ++dt)
#pragma unroll
      for (int r = 0; r < 8; ++r) acc[dt][r] *= alpha[r];
    // C-layout (lane=col, vgpr=row) -> A-layout via per-wave LDS tile.
#pragma unroll
    for (int r = 0; r < 8; ++r) {
      pl[(rb + r) * 32 + colL]      = f2bf(p0[r]);
      pl[(rb + r) * 32 + 16 + colL] = f2bf(p1[r]);
    }
    asm volatile("" ::: "memory");       // same-wave DS ops are HW in-order
    v16bf pa = load_a_frag(pl, 32);
    asm volatile("" ::: "memory");
#pragma unroll
    for (int dt = 0; dt < 4; ++dt) {
      v16bf vb = load_b_frag(&Vtile[dt * 16][0], 32);
      acc[dt] = WMMA_BF16(pa, vb, acc[dt]);
    }
  }

#pragma unroll
  for (int r = 0; r < 8; ++r) lrow[r] = 1.0f / lrow[r];
#pragma unroll
  for (int dt = 0; dt < 4; ++dt) {
#pragma unroll
    for (int r = 0; r < 8; ++r) {
      const int m    = mtile * 16 + rb + r;
      const int dcol = dt * 16 + colL;
      Out[((size_t)b * N_ + m) * D_ + h * HD_ + dcol] = f2bf(acc[dt][r] * lrow[r]);
    }
  }
}

// ---------------- kernel 4: output projection + bias + residual (32x64 tile) ----------------
__global__ void k_oproj(const bf16_t* __restrict__ ab,
                        const bf16_t* __restrict__ wo,
                        const float* __restrict__ bo,
                        const float* __restrict__ xres,
                        float* __restrict__ y) {
  const int w = blockIdx.x * 8 + (threadIdx.x >> 5);
  const int mbase = (w >> 3) << 5;
  const int nbase = (w & 7) << 6;

  v8f acc0[4], acc1[4];
#pragma unroll
  for (int s = 0; s < 4; ++s) { acc0[s] = v8zero(); acc1[s] = v8zero(); }

  for (int k = 0; k < D_; k += 32) {
    __builtin_prefetch(ab + (size_t)mbase * D_ + k + 32, 0, 0);
    v16bf a0 = load_a_frag(ab + (size_t)mbase * D_ + k, D_);
    v16bf a1 = load_a_frag(ab + (size_t)(mbase + 16) * D_ + k, D_);
#pragma unroll
    for (int s = 0; s < 4; ++s) {
      v16bf b = load_b_frag(wo + (size_t)(nbase + s * 16) * D_ + k, D_);
      acc0[s] = WMMA_BF16(a0, b, acc0[s]);
      acc1[s] = WMMA_BF16(a1, b, acc1[s]);
    }
  }

  const int l    = threadIdx.x & 31;
  const int colL = l & 15;
  const int rb   = (l >> 4) << 3;
#pragma unroll
  for (int half = 0; half < 2; ++half) {
#pragma unroll
    for (int s = 0; s < 4; ++s) {
      const int n = nbase + s * 16 + colL;
      const float bn = bo[n];
#pragma unroll
      for (int r = 0; r < 8; ++r) {
        const int m = mbase + half * 16 + rb + r;
        y[(size_t)m * D_ + n] =
            (half ? acc1[s][r] : acc0[s][r]) + bn + xres[(size_t)m * D_ + n];
      }
    }
  }
}

// ---------------- kernel 5: layernorm ----------------
__global__ void k_ln(const float* __restrict__ y,
                     const float* __restrict__ gamma,
                     const float* __restrict__ beta,
                     float* __restrict__ out) {
  __shared__ float ssum[4], ssq[4];
  const int row = blockIdx.x;
  const int t   = threadIdx.x;            // 0..127
  const float* x = y + (size_t)row * D_;
  float4 v = *(const float4*)(x + t * 4);
  float s  = v.x + v.y + v.z + v.w;
  float sq = v.x * v.x + v.y * v.y + v.z * v.z + v.w * v.w;
#pragma unroll
  for (int m = 1; m < 32; m <<= 1) {
    s  += __shfl_xor(s,  m, 32);
    sq += __shfl_xor(sq, m, 32);
  }
  const int wv = t >> 5;
  if ((t & 31) == 0) { ssum[wv] = s; ssq[wv] = sq; }
  __syncthreads();
  s  = ssum[0] + ssum[1] + ssum[2] + ssum[3];
  sq = ssq[0]  + ssq[1]  + ssq[2]  + ssq[3];
  const float mu  = s * (1.0f / D_);
  const float var = sq * (1.0f / D_) - mu * mu;   // population variance (jnp.var)
  const float rs  = rsqrtf(var + 1e-5f);
  float4 g  = *(const float4*)(gamma + t * 4);
  float4 bb = *(const float4*)(beta + t * 4);
  float4 o;
  o.x = (v.x - mu) * rs * g.x + bb.x;
  o.y = (v.y - mu) * rs * g.y + bb.y;
  o.z = (v.z - mu) * rs * g.z + bb.z;
  o.w = (v.w - mu) * rs * g.w + bb.w;
  *(float4*)(out + (size_t)row * D_ + t * 4) = o;
}

// ---------------- launcher ----------------
extern "C" void kernel_launch(void* const* d_in, const int* in_sizes, int n_in,
                              void* d_out, int out_size, void* d_ws, size_t ws_size,
                              hipStream_t stream) {
  (void)in_sizes; (void)n_in; (void)out_size; (void)ws_size;
  const float* nf    = (const float*)d_in[0];
  const float* adj   = (const float*)d_in[1];
  const float* Wq    = (const float*)d_in[2];
  const float* bq    = (const float*)d_in[3];
  const float* Wk    = (const float*)d_in[4];
  const float* bk    = (const float*)d_in[5];
  const float* Wv    = (const float*)d_in[6];
  const float* bv    = (const float*)d_in[7];
  const float* Wo    = (const float*)d_in[8];
  const float* bo    = (const float*)d_in[9];
  const float* gamma = (const float*)d_in[10];
  const float* beta  = (const float*)d_in[11];
  float* out = (float*)d_out;

  char* ws = (char*)d_ws;
  const size_t nD = (size_t)B_ * N_ * D_;   // 4,194,304 elements
  const size_t wE = (size_t)D_ * D_;        // 262,144 elements
  size_t off = 0;
  bf16_t* xb  = (bf16_t*)(ws + off); off += nD * 2;
  bf16_t* wqb = (bf16_t*)(ws + off); off += wE * 2;
  bf16_t* wkb = (bf16_t*)(ws + off); off += wE * 2;
  bf16_t* wvb = (bf16_t*)(ws + off); off += wE * 2;
  bf16_t* wob = (bf16_t*)(ws + off); off += wE * 2;
  bf16_t* Qb  = (bf16_t*)(ws + off); off += nD * 2;
  bf16_t* Kb  = (bf16_t*)(ws + off); off += nD * 2;
  bf16_t* Vtb = (bf16_t*)(ws + off); off += nD * 2;
  bf16_t* Ab  = (bf16_t*)(ws + off); off += nD * 2;
  float*  yr  = (float*)(ws + off);  off += nD * 4;

  k_cvt_bf16<<<(int)(nD / 4 / 256), 256, 0, stream>>>(nf, xb, (int)nD);
  k_cvt_bf16<<<(int)(wE / 4 / 256), 256, 0, stream>>>(Wq, wqb, (int)wE);
  k_cvt_bf16<<<(int)(wE / 4 / 256), 256, 0, stream>>>(Wk, wkb, (int)wE);
  k_cvt_bf16<<<(int)(wE / 4 / 256), 256, 0, stream>>>(Wv, wvb, (int)wE);
  k_cvt_bf16<<<(int)(wE / 4 / 256), 256, 0, stream>>>(Wo, wob, (int)wE);

  // 3 mats * 256 row-tiles(32) * 8 col-tiles = 6144 waves / 8 per block
  k_qkv<<<768, 256, 0, stream>>>(xb, wqb, wkb, wvb, bq, bk, bv, Qb, Kb, Vtb);
  // 8*8*64 = 4096 waves / 8 per block (8 waves share (b,h))
  k_attn<<<512, 256, 0, stream>>>(Qb, Kb, Vtb, adj, Ab);
  // 256 row-tiles(32) * 8 col-tiles = 2048 waves / 8 per block
  k_oproj<<<256, 256, 0, stream>>>(Ab, wob, bo, nf, yr);
  k_ln<<<B_ * N_, 128, 0, stream>>>(yr, gamma, beta, out);
}
